// GlobalGBST_84988812853375
// MI455X (gfx1250) — matrix operand order repaired and checked by reading, not compiled, not run on hardware
//
#include <hip/hip_runtime.h>
#include <hip/hip_bf16.h>
#include <float.h>
#include <stdint.h>

typedef __attribute__((ext_vector_type(16))) __bf16   v16bf;
typedef __attribute__((ext_vector_type(8)))  float    v8f;
typedef __attribute__((ext_vector_type(4)))  uint32_t u32x4;
typedef __attribute__((ext_vector_type(8)))  uint32_t u32x8;

#define BB 8
#define LL 2048
#define DD 256
#define NL 9
#define PADTOK 0
#define GTOT 6145   // 1024+1024+683*3+512*4 compact group slots per batch

__device__ __constant__ int c_sizes[NL] = {2,2,3,3,3,4,4,4,4};
__device__ __constant__ int c_ng[NL]    = {1024,1024,683,683,683,512,512,512,512};
__device__ __constant__ int c_off[NL]   = {0,1024,2048,2731,3414,4097,4609,5121,5633};

// ---------------------------------------------------------------------------
// K1: embedding*sqrt(D) -> depthwise conv4 (end-padded) -> proj GEMM (+bias),
//     zero rows where sequence==PAD.  GEMM via v_wmma_f32_16x16x32_bf16.
// Block: 16 rows x 256 cols of output.  8 waves, 2 N-tiles each, K=256 loop.
// ---------------------------------------------------------------------------
__global__ __launch_bounds__(256) void k_embed_conv_proj(
    const int*   __restrict__ seq,
    const float* __restrict__ emb,
    const float* __restrict__ conv_w,   // [D,1,4]
    const float* __restrict__ conv_b,
    const float* __restrict__ proj_w,   // [out=D, in=D]
    const float* __restrict__ proj_b,
    float*       __restrict__ xbuf)     // [B*L, D]
{
  __shared__ __bf16 As[16][DD + 8];
  __shared__ float  padf[16];           // 1.0 = keep row, 0.0 = PAD row
  const int t    = threadIdx.x;
  const int row0 = blockIdx.x * 16;     // L=2048 multiple of 16 -> no batch straddle

  // ---- stage A: conv activations for 16 rows x 256 features, bf16 in LDS ----
  {
    const int m     = t >> 4;           // 0..15 (row in tile)
    const int dbase = (t & 15) * 16;    // 16 features per thread
    const int row   = row0 + m;
    const int b = row / LL, l = row % LL;
    int s4[4];
#pragma unroll
    for (int j = 0; j < 4; ++j) {
      const int lj = l + j;
      s4[j] = (lj < LL) ? seq[b * LL + lj] : -1;   // -1 => zero-pad beyond seq end
    }
    if ((t & 15) == 0) padf[m] = (s4[0] == PADTOK) ? 0.0f : 1.0f;
#pragma unroll
    for (int dd = 0; dd < 16; ++dd) {
      const int d = dbase + dd;
      float acc = conv_b[d];
#pragma unroll
      for (int j = 0; j < 4; ++j)
        if (s4[j] >= 0)
          acc += 16.0f * emb[s4[j] * DD + d] * conv_w[d * 4 + j];  // sqrt(256)=16
      As[m][d] = (__bf16)acc;
    }
  }
  __syncthreads();

  const int wave = t >> 5;
  const int lane = t & 31;
  const int half = lane >> 4;           // 0: lanes 0-15, 1: lanes 16-31
  const int l16  = lane & 15;
  const int n0   = (wave * 2) * 16 + l16;
  const int n1   = n0 + 16;

  v8f c0 = {}; v8f c1 = {};
  for (int ks = 0; ks < DD / 32; ++ks) {
    const int k0 = ks * 32;
    // A fragment (16-bit A 16x32 layout, ISA 7.12.2)
    v16bf a;
    {
      const int kb1 = k0 + (half ? 8 : 0);        // VGPR0-3: K kb1..kb1+7
      const int kb2 = k0 + 16 + (half ? 8 : 0);   // VGPR4-7: K kb2..kb2+7
#pragma unroll
      for (int i = 0; i < 8; ++i) a[i]     = As[l16][kb1 + i];
#pragma unroll
      for (int i = 0; i < 8; ++i) a[8 + i] = As[l16][kb2 + i];
    }
    // B fragments: B[k][n] = proj_w[n][k] -> contiguous along k per lane
    v16bf b0, b1;
    const int kbB = k0 + (half ? 8 : 0);
#pragma unroll
    for (int i = 0; i < 16; ++i) {
      b0[i] = (__bf16)proj_w[n0 * DD + kbB + i + (i & 8)];
      b1[i] = (__bf16)proj_w[n1 * DD + kbB + i + (i & 8)];
    }
    c0 = __builtin_amdgcn_wmma_f32_16x16x32_bf16(false, a, false, b0, (short)0, c0, false, false);
    c1 = __builtin_amdgcn_wmma_f32_16x16x32_bf16(false, a, false, b1, (short)0, c1, false, false);
  }

  // epilogue: (+proj_b) * pad-mask, branch-free (C layout: VGPR r -> M=r / r+8)
  const float pb0 = proj_b[n0];
  const float pb1 = proj_b[n1];
#pragma unroll
  for (int r = 0; r < 8; ++r) {
    const int m   = r + (half ? 8 : 0);
    const int row = row0 + m;
    const float keep = padf[m];
    xbuf[(size_t)row * DD + n0] = (c0[r] + pb0) * keep;
    xbuf[(size_t)row * DD + n1] = (c1[r] + pb1) * keep;
  }
}

// ---------------------------------------------------------------------------
// K2: segment means per (b, layer, group); compact layout [B][GTOT][D].
// ---------------------------------------------------------------------------
__global__ __launch_bounds__(256) void k_pool_means(
    const int*   __restrict__ gid,      // [B, NL, L]
    const float* __restrict__ xbuf,     // [B*L, D]
    float*       __restrict__ means)    // [B, GTOT, D]
{
  const int blk = blockIdx.x;
  const int g   = blk & 1023;
  const int li  = (blk >> 10) % NL;
  const int b   = blk / (NL * 1024);
  if (g >= c_ng[li]) return;
  const int s    = c_sizes[li];
  const int p0   = g * s;
  const int pend = min(p0 + s, LL);
  const int* grow = gid + ((size_t)b * NL + li) * LL;

  int cnt = 0;
  for (int p = p0; p < pend; ++p) cnt += (grow[p] == g + 1);
  const float inv = cnt ? (1.0f / (float)cnt) : 0.0f;

  const int d = threadIdx.x;
  float sum = 0.0f;
  for (int p = p0; p < pend; ++p) {
    const float keep = (grow[p] == g + 1) ? 1.0f : 0.0f;
    sum += keep * xbuf[((size_t)b * LL + p) * DD + d];
  }
  means[((size_t)b * GTOT + c_off[li] + g) * DD + d] = sum * inv;
}

// ---------------------------------------------------------------------------
// K3: per position: 10 scores (dot with score_w), masked softmax, weighted mix.
// One block (256 threads == D) per position.
// ---------------------------------------------------------------------------
__global__ __launch_bounds__(256) void k_mix_softmax(
    const int*   __restrict__ seq,
    const int*   __restrict__ gid,
    const float* __restrict__ xbuf,
    const float* __restrict__ means,
    const float* __restrict__ score_w,
    const float* __restrict__ score_b,  // scalar
    float*       __restrict__ out)      // [B*L, D]  (pre-FFN y)
{
  __shared__ float red[256];
  __shared__ float sc[10];
  const int pos = blockIdx.x;           // b*L + l
  const int b = pos / LL, l = pos % LL;
  const int d = threadIdx.x;

  float rep[10];
  bool  msk[10];
  rep[0] = xbuf[(size_t)pos * DD + d];
  msk[0] = (seq[pos] == PADTOK);
#pragma unroll
  for (int li = 0; li < NL; ++li) {
    const int g = gid[((size_t)b * NL + li) * LL + l];
    msk[li + 1] = (g == PADTOK);
    rep[li + 1] = (g == PADTOK) ? 0.0f
                : means[((size_t)b * GTOT + c_off[li] + (g - 1)) * DD + d];
  }

  const float sw = score_w[d];
  for (int s = 0; s < 10; ++s) {
    red[d] = rep[s] * sw;
    __syncthreads();
    for (int st = 128; st > 0; st >>= 1) {
      if (d < st) red[d] += red[d + st];
      __syncthreads();
    }
    if (d == 0) sc[s] = red[0] + score_b[0];
    __syncthreads();
  }

  // masked softmax over 10 (replicated per thread); all-masked -> uniform 1/10
  float w[10];
  float mx = -FLT_MAX;
#pragma unroll
  for (int s = 0; s < 10; ++s) {
    w[s] = msk[s] ? -FLT_MAX : sc[s];
    mx = fmaxf(mx, w[s]);
  }
  float den = 0.0f;
#pragma unroll
  for (int s = 0; s < 10; ++s) { w[s] = __expf(w[s] - mx); den += w[s]; }
  const float inv = 1.0f / den;
  float acc = 0.0f;
#pragma unroll
  for (int s = 0; s < 10; ++s) acc += rep[s] * w[s];
  out[(size_t)pos * DD + d] = acc * inv;
}

// ---------------------------------------------------------------------------
// K4: y += relu(y @ ff_w^T + ff_b), WMMA bf16 GEMM, in-place on d_out.
// A-tile (16 rows x 256 f32, contiguous) staged to LDS by the Tensor Data
// Mover: one tensor_load_to_lds per block (wave 0), TENSORcnt-tracked,
// s_wait_tensorcnt 0 + workgroup barrier before use (ISA 08_async_tensor §8).
// f32->bf16 conversion happens at fragment-build time; residual read from LDS.
// ---------------------------------------------------------------------------
__global__ __launch_bounds__(256) void k_ffn_residual(
    const float* __restrict__ ff_w,     // [out=D, in=D]
    const float* __restrict__ ff_b,
    float*       __restrict__ y)        // [B*L, D] in/out
{
  __shared__ float Asf[16][DD];         // 16 KB, TDM destination (contiguous)
  const int t    = threadIdx.x;
  const int row0 = blockIdx.x * 16;

  if ((t >> 5) == 0) {                  // wave 0 issues the TDM descriptor
    const unsigned long long ga =
        (unsigned long long)(uintptr_t)(y + (size_t)row0 * DD);
    const uint32_t lds = (uint32_t)(uintptr_t)&Asf[0][0]; // low 32b = LDS offset
    // D# group 0: count=1 | lds_addr | global_addr[56:0] | type=2 (bits 127:126)
    u32x4 g0 = { 1u, lds, (uint32_t)ga,
                 ((uint32_t)(ga >> 32) & 0x01FFFFFFu) | (2u << 30) };
    // D# group 1: data_size=4B(code 2); tensor_dim0=256; tensor_dim1=B*L;
    //             tile_dim0=256; tile_dim1=16; tensor_dim0_stride=256
    u32x8 g1 = { (2u << 16),                    // workgroup_mask=0, data_size=2
                 (256u & 0xFFFFu) << 16,        // tensor_dim0[15:0] @ bits 63:48
                 ((uint32_t)(BB * LL) & 0xFFFFu) << 16, // tensor_dim1[15:0] @ 95:80
                 (256u << 16),                  // tile_dim0 @ bits 127:112
                 16u,                           // tile_dim1 @ bits 143:128
                 256u,                          // tensor_dim0_stride[31:0]
                 0u, 0u };
    asm volatile("tensor_load_to_lds %0, %1" :: "s"(g0), "s"(g1) : "memory");
    __builtin_amdgcn_s_wait_tensorcnt(0);
  }
  __syncthreads();

  const int wave = t >> 5;
  const int lane = t & 31;
  const int half = lane >> 4;
  const int l16  = lane & 15;
  const int n0   = (wave * 2) * 16 + l16;
  const int n1   = n0 + 16;

  v8f c0 = {}; v8f c1 = {};
  for (int ks = 0; ks < DD / 32; ++ks) {
    const int k0 = ks * 32;
    v16bf a;
    {
      const int kb1 = k0 + (half ? 8 : 0);
      const int kb2 = k0 + 16 + (half ? 8 : 0);
#pragma unroll
      for (int i = 0; i < 8; ++i) a[i]     = (__bf16)Asf[l16][kb1 + i];
#pragma unroll
      for (int i = 0; i < 8; ++i) a[8 + i] = (__bf16)Asf[l16][kb2 + i];
    }
    v16bf b0, b1;
    const int kbB = k0 + (half ? 8 : 0);
#pragma unroll
    for (int i = 0; i < 16; ++i) {
      b0[i] = (__bf16)ff_w[n0 * DD + kbB + i + (i & 8)];
      b1[i] = (__bf16)ff_w[n1 * DD + kbB + i + (i & 8)];
    }
    c0 = __builtin_amdgcn_wmma_f32_16x16x32_bf16(false, a, false, b0, (short)0, c0, false, false);
    c1 = __builtin_amdgcn_wmma_f32_16x16x32_bf16(false, a, false, b1, (short)0, c1, false, false);
  }

  const float fb0 = ff_b[n0];
  const float fb1 = ff_b[n1];
#pragma unroll
  for (int r = 0; r < 8; ++r) {
    const int m   = r + (half ? 8 : 0);
    const size_t row = (size_t)(row0 + m);
    const float y0 = Asf[m][n0];        // residual from LDS copy
    const float y1 = Asf[m][n1];
    y[row * DD + n0] = y0 + fmaxf(0.0f, c0[r] + fb0);
    y[row * DD + n1] = y1 + fmaxf(0.0f, c1[r] + fb1);
  }
}

// ---------------------------------------------------------------------------
extern "C" void kernel_launch(void* const* d_in, const int* in_sizes, int n_in,
                              void* d_out, int out_size, void* d_ws, size_t ws_size,
                              hipStream_t stream)
{
  (void)in_sizes; (void)n_in; (void)out_size; (void)ws_size;
  const int*   seq     = (const int*)  d_in[0];
  const int*   gid     = (const int*)  d_in[1];
  const float* emb     = (const float*)d_in[2];
  const float* conv_w  = (const float*)d_in[3];
  const float* conv_b  = (const float*)d_in[4];
  const float* proj_w  = (const float*)d_in[5];
  const float* proj_b  = (const float*)d_in[6];
  const float* score_w = (const float*)d_in[7];
  const float* score_b = (const float*)d_in[8];
  const float* ff_w    = (const float*)d_in[9];
  const float* ff_b    = (const float*)d_in[10];
  float* out = (float*)d_out;

  float* xbuf  = (float*)d_ws;                           // B*L*D f32 (16 MB)
  float* means = xbuf + (size_t)BB * LL * DD;            // B*GTOT*D f32 (~50 MB)

  k_embed_conv_proj<<<(BB * LL) / 16, 256, 0, stream>>>(
      seq, emb, conv_w, conv_b, proj_w, proj_b, xbuf);
  k_pool_means<<<BB * NL * 1024, 256, 0, stream>>>(gid, xbuf, means);
  k_mix_softmax<<<BB * LL, 256, 0, stream>>>(
      seq, gid, xbuf, means, score_w, score_b, out);
  k_ffn_residual<<<(BB * LL) / 16, 256, 0, stream>>>(ff_w, ff_b, out);
}
